// AttentionBlock_6648609374323
// MI455X (gfx1250) — matrix operand order repaired and checked by reading, not compiled, not run on hardware
//
#include <hip/hip_runtime.h>

// ---------------------------------------------------------------------------
// Types for CDNA5 WMMA (wave32, 16x16x32 bf16 -> f32)
// ---------------------------------------------------------------------------
typedef __attribute__((ext_vector_type(16))) __bf16        v16bf;
typedef __attribute__((ext_vector_type(8)))  float         v8f;
typedef __attribute__((ext_vector_type(4)))  unsigned int  u32x4;

union FragU { u32x4 q[2]; v16bf v; };

__device__ __forceinline__ unsigned short f2bf(float f) {
  union { float f; unsigned int u; } c; c.f = f;
  return (unsigned short)((c.u + 0x7FFFu + ((c.u >> 16) & 1u)) >> 16); // RNE
}
__device__ __forceinline__ float bf2f(unsigned short h) {
  union { unsigned int u; float f; } c; c.u = ((unsigned int)h) << 16; return c.f;
}

// A-fragment (16x32 bf16): lane holds row m=lane%16, K = kh*8..+7 and 16+kh*8..+7
// p must already point at  row_base + kh*8  (16B aligned)
__device__ __forceinline__ v16bf load_a_frag(const unsigned short* p) {
  FragU u; u.q[0] = *(const u32x4*)p; u.q[1] = *(const u32x4*)(p + 16); return u.v;
}
// B-fragment (32x16 bf16, stored as B^T [n][k]): lane holds column n=lane%16,
// K = kh*8..+7 and 16+kh*8..+7.  p -> col_base + kh*8
__device__ __forceinline__ v16bf load_b_frag(const unsigned short* p) {
  FragU u; u.q[0] = *(const u32x4*)p; u.q[1] = *(const u32x4*)(p + 16); return u.v;
}

__device__ __forceinline__ v8f wmma_bf16(v16bf a, v16bf b, v8f c) {
  return __builtin_amdgcn_wmma_f32_16x16x32_bf16(false, a, false, b, (short)0, c,
                                                 false, false);
}

// ---------------------------------------------------------------------------
// Transposing fp32 -> bf16 cast:  in [K,N] f32  ->  out [N,K] bf16
// LDS-tiled so both global streams are coalesced.
// ---------------------------------------------------------------------------
__global__ __launch_bounds__(256) void transpose_cast_kernel(
    const float* __restrict__ in, unsigned short* __restrict__ out, int K, int N) {
  __shared__ float tile[32][33];
  const int n0 = blockIdx.x * 32, k0 = blockIdx.y * 32;
  const int tx = threadIdx.x, ty = threadIdx.y;   // 32 x 8
#pragma unroll
  for (int j = 0; j < 4; ++j) {
    int k = k0 + ty + j * 8, n = n0 + tx;
    tile[ty + j * 8][tx] = (k < K && n < N) ? in[(size_t)k * N + n] : 0.f;
  }
  __syncthreads();
#pragma unroll
  for (int j = 0; j < 4; ++j) {
    int n = n0 + ty + j * 8, k = k0 + tx;
    if (n < N && k < K) out[(size_t)n * K + k] = f2bf(tile[tx][ty + j * 8]);
  }
}

// ---------------------------------------------------------------------------
// RMSNorm: h1 = bf16(rmsnorm(x)*g1), h2 = bf16(rmsnorm(x)*g2), out = x
// ---------------------------------------------------------------------------
__global__ __launch_bounds__(256) void rmsnorm_dual_kernel(
    const float* __restrict__ x, const float* __restrict__ g1,
    const float* __restrict__ g2, unsigned short* __restrict__ h1,
    unsigned short* __restrict__ h2, float* __restrict__ out, int C) {
  const int row = blockIdx.x;
  const float* xr = x + (size_t)row * C;
  float ss = 0.f;
  for (int c = threadIdx.x; c < C; c += 256) { float v = xr[c]; ss += v * v; }
#pragma unroll
  for (int m = 16; m >= 1; m >>= 1) ss += __shfl_xor(ss, m, 32);
  __shared__ float red[8];
  __shared__ float snorm;
  if ((threadIdx.x & 31) == 0) red[threadIdx.x >> 5] = ss;
  __syncthreads();
  if (threadIdx.x == 0) {
    float t = 0.f;
    for (int i = 0; i < 8; ++i) t += red[i];
    snorm = rsqrtf(t / (float)C + 1e-6f);
  }
  __syncthreads();
  const float nrm = snorm;
  for (int c = threadIdx.x; c < C; c += 256) {
    float v = xr[c];
    float vn = v * nrm;
    h1[(size_t)row * C + c] = f2bf(vn * g1[c]);
    h2[(size_t)row * C + c] = f2bf(vn * g2[c]);
    out[(size_t)row * C + c] = v;   // residual seed: out = x
  }
}

// ---------------------------------------------------------------------------
// bf16 GEMM with pre-transposed B:  C[M,N] (+)= A[M,K] * Bt[N,K]^T
// A row-major [M,K], Bt row-major [N,K] (i.e. original B transposed).
// EPI==0: store bf16 C.  EPI==1: fp32 C += acc.
// Block tile 128x128x32, 8 waves (wave32), wave tile 32x64 = 2x4 WMMA tiles.
// Both LDS tiles stage as straight b128 copies (no scatter); double-buffered;
// global_prefetch for the k+2 tile.
// ---------------------------------------------------------------------------
template <int EPI>
__global__ __launch_bounds__(256) void gemm_bf16_kernel(
    const unsigned short* __restrict__ A, const unsigned short* __restrict__ Bt,
    void* __restrict__ Cp, int M, int N, int K) {
  constexpr int BMc = 128, BNc = 128, BKc = 32, LD = 40; // pad, 16B aligned
  __shared__ unsigned short As[2][BMc * LD];
  __shared__ unsigned short Bs[2][BNc * LD];

  const int tid  = threadIdx.x;
  const int wave = tid >> 5, lane = tid & 31;
  const int lm = lane & 15, kh = lane >> 4;
  const int wm = wave & 3, wn = wave >> 2;
  const int m0 = blockIdx.y * BMc, n0 = blockIdx.x * BNc;
  const bool kvec = ((K & 7) == 0);   // 16B-aligned rows (uniform)

  const v8f vzero = {0.f, 0.f, 0.f, 0.f, 0.f, 0.f, 0.f, 0.f};
  v8f acc[2][4];
#pragma unroll
  for (int i = 0; i < 2; ++i)
#pragma unroll
    for (int j = 0; j < 4; ++j) acc[i][j] = vzero;

  // Stage a 128x32 slab of a row-major [rows,K] matrix into LDS (ld=LD).
  auto stage_mat = [&](const unsigned short* __restrict__ src_mat,
                       unsigned short* __restrict__ dstbuf, int rows, int r0,
                       int k0) {
    const bool fast = kvec && (k0 + BKc <= K);
#pragma unroll
    for (int p = 0; p < 2; ++p) {
      int c = p * 256 + tid;
      int row = c >> 2;
      int kc  = (c & 3) << 3;
      int gr  = r0 + row;
      const unsigned short* src = src_mat + (size_t)gr * K + (size_t)(k0 + kc);
      unsigned short* dst = &dstbuf[row * LD + kc];
      if (fast) {
        u32x4 v = {0u, 0u, 0u, 0u};
        if (gr < rows) v = *(const u32x4*)src;
        *(u32x4*)dst = v;
      } else {
#pragma unroll
        for (int j = 0; j < 4; ++j) {
          unsigned int v = 0u;
          if (gr < rows && (k0 + kc + 2 * j) < K)
            v = *(const unsigned int*)(src + 2 * j);
          *(unsigned int*)(dst + 2 * j) = v;
        }
      }
    }
  };
  auto stage = [&](int buf, int k0) {
    stage_mat(A,  As[buf], M, m0, k0);
    stage_mat(Bt, Bs[buf], N, n0, k0);
  };

  const int nk = (K + BKc - 1) / BKc;
  stage(0, 0);
  for (int kt = 0; kt < nk; ++kt) {
    __syncthreads();
    if (kt + 1 < nk) stage((kt + 1) & 1, (kt + 1) * BKc);
    if (kt + 2 < nk) {
      // pull the k+2 tile toward cache while double-buffer hides k+1
      const int kp = (kt + 2) * BKc;
      const int row = tid >> 1;                  // 0..127, each row twice
      if (m0 + row < M) __builtin_prefetch(A  + (size_t)(m0 + row) * K + kp, 0, 1);
      if (n0 + row < N) __builtin_prefetch(Bt + (size_t)(n0 + row) * K + kp, 0, 1);
    }
    const int cur = kt & 1;
    v16bf af[2], bf[4];
#pragma unroll
    for (int ti = 0; ti < 2; ++ti)
      af[ti] = load_a_frag(&As[cur][(wm * 32 + ti * 16 + lm) * LD + kh * 8]);
#pragma unroll
    for (int tj = 0; tj < 4; ++tj)
      bf[tj] = load_b_frag(&Bs[cur][(wn * 64 + tj * 16 + lm) * LD + kh * 8]);
#pragma unroll
    for (int ti = 0; ti < 2; ++ti)
#pragma unroll
      for (int tj = 0; tj < 4; ++tj)
        acc[ti][tj] = wmma_bf16(af[ti], bf[tj], acc[ti][tj]);
  }

  // Epilogue: C layout -> row = kh*8 + r, col = lm
#pragma unroll
  for (int ti = 0; ti < 2; ++ti) {
#pragma unroll
    for (int tj = 0; tj < 4; ++tj) {
      int col = n0 + wn * 64 + tj * 16 + lm;
#pragma unroll
      for (int r = 0; r < 8; ++r) {
        int row = m0 + wm * 32 + ti * 16 + kh * 8 + r;
        if (row < M && col < N) {
          size_t idx = (size_t)row * N + col;
          if constexpr (EPI == 0)
            ((unsigned short*)Cp)[idx] = f2bf(acc[ti][tj][r]);
          else
            ((float*)Cp)[idx] += acc[ti][tj][r];
        }
      }
    }
  }
}

// ---------------------------------------------------------------------------
// Flash attention (causal), bf16 WMMA, fp32 softmax state.
// Block: 128 threads = 4 waves; block handles 64 q rows of one (b,h);
// each wave owns 16 q rows. qkv layout: [B*T, 3C], q|k|v split along cols.
// ---------------------------------------------------------------------------
__global__ __launch_bounds__(128) void attn_kernel(
    const unsigned short* __restrict__ qkv, unsigned short* __restrict__ y,
    int T, int C, int H) {
  const int hd = C / H;          // 64
  const int stride = 3 * C;      // 3072
  const int qt = blockIdx.x;
  const int b = blockIdx.y / H;
  const int h = blockIdx.y % H;
  const int tid = threadIdx.x;
  const int wave = tid >> 5, lane = tid & 31;
  const int lm = lane & 15, kh = lane >> 4;

  __shared__ unsigned short Vt[64 * 72];      // V tile transposed [d][t]
  __shared__ unsigned short Ps[4][16 * 72];   // per-wave P (bf16)

  const size_t base = (size_t)b * T * stride;
  const int q0 = qt * 64;
  const int qrow = q0 + wave * 16 + lm;       // A-frag row for this lane

  // Q fragments (K-dim = hd = 64 -> 2 WMMA k-steps)
  FragU aq[2];
  {
    const unsigned short* qp = qkv + base + (size_t)qrow * stride + h * hd;
#pragma unroll
    for (int ks = 0; ks < 2; ++ks) {
      const unsigned short* p = qp + ks * 32 + kh * 8;
      aq[ks].q[0] = *(const u32x4*)p;
      aq[ks].q[1] = *(const u32x4*)(p + 16);
    }
  }

  const v8f vzero = {0.f, 0.f, 0.f, 0.f, 0.f, 0.f, 0.f, 0.f};
  float Mrow[8], Lrow[8];
  v8f O[4];
#pragma unroll
  for (int r = 0; r < 8; ++r) { Mrow[r] = -3.0e38f; Lrow[r] = 0.f; }
#pragma unroll
  for (int dj = 0; dj < 4; ++dj) O[dj] = vzero;

  for (int kt = 0; kt <= qt; ++kt) {
    const int kv0 = kt * 64;

    // stage V tile transposed into LDS: Vt[d][t]
#pragma unroll
    for (int p4 = 0; p4 < 4; ++p4) {
      int c = p4 * 128 + tid;             // 512 chunks of 8 along d
      int tl = c >> 3;
      int dc = (c & 7) << 3;
      const unsigned short* vp =
          qkv + base + (size_t)(kv0 + tl) * stride + 2 * C + h * hd + dc;
      u32x4 v = *(const u32x4*)vp;
#pragma unroll
      for (int j = 0; j < 4; ++j) {
        unsigned int w = v[j];
        Vt[(dc + 2 * j + 0) * 72 + tl] = (unsigned short)(w & 0xffffu);
        Vt[(dc + 2 * j + 1) * 72 + tl] = (unsigned short)(w >> 16);
      }
    }
    __syncthreads();

    // S = Q @ K^T  (B-frag of K^T == K's natural row-major layout)
    v8f s[4];
#pragma unroll
    for (int tj = 0; tj < 4; ++tj) s[tj] = vzero;
#pragma unroll
    for (int ks = 0; ks < 2; ++ks) {
#pragma unroll
      for (int tj = 0; tj < 4; ++tj) {
        const unsigned short* kp = qkv + base +
            (size_t)(kv0 + tj * 16 + lm) * stride + C + h * hd + ks * 32 + kh * 8;
        FragU bk;
        bk.q[0] = *(const u32x4*)kp;
        bk.q[1] = *(const u32x4*)(kp + 16);
        s[tj] = wmma_bf16(aq[ks].v, bk.v, s[tj]);
      }
    }

    // causal mask + scale + online softmax (rows live in 16-lane halves)
    float mnew[8];
#pragma unroll
    for (int r = 0; r < 8; ++r) {
      const int q = q0 + wave * 16 + kh * 8 + r;
      float m = -3.0e38f;
#pragma unroll
      for (int tj = 0; tj < 4; ++tj) {
        int t = kv0 + tj * 16 + lm;
        float v = s[tj][r] * 0.125f;     // 1/sqrt(hd)
        v = (t <= q) ? v : -3.0e38f;
        s[tj][r] = v;
        m = fmaxf(m, v);
      }
      m = fmaxf(m, __shfl_xor(m, 1, 32));
      m = fmaxf(m, __shfl_xor(m, 2, 32));
      m = fmaxf(m, __shfl_xor(m, 4, 32));
      m = fmaxf(m, __shfl_xor(m, 8, 32));
      mnew[r] = fmaxf(Mrow[r], m);
      float sum = 0.f;
#pragma unroll
      for (int tj = 0; tj < 4; ++tj) {
        float e = __expf(s[tj][r] - mnew[r]);
        s[tj][r] = e;
        sum += e;
      }
      sum += __shfl_xor(sum, 1, 32);
      sum += __shfl_xor(sum, 2, 32);
      sum += __shfl_xor(sum, 4, 32);
      sum += __shfl_xor(sum, 8, 32);
      float corr = __expf(Mrow[r] - mnew[r]);
      Lrow[r] = Lrow[r] * corr + sum;
      Mrow[r] = mnew[r];
#pragma unroll
      for (int dj = 0; dj < 4; ++dj) O[dj][r] *= corr;
    }

    // write P (C-layout -> row-major bf16 in LDS) for A-frag reload
#pragma unroll
    for (int r = 0; r < 8; ++r)
#pragma unroll
      for (int tj = 0; tj < 4; ++tj)
        Ps[wave][(kh * 8 + r) * 72 + tj * 16 + lm] = f2bf(s[tj][r]);
    __syncthreads();

    // O += P @ V
#pragma unroll
    for (int ks = 0; ks < 2; ++ks) {
      const unsigned short* pp = &Ps[wave][lm * 72 + ks * 32 + kh * 8];
      FragU ap;
      ap.q[0] = *(const u32x4*)pp;
      ap.q[1] = *(const u32x4*)(pp + 16);
#pragma unroll
      for (int dj = 0; dj < 4; ++dj) {
        const unsigned short* vp = &Vt[(dj * 16 + lm) * 72 + ks * 32 + kh * 8];
        FragU bv;
        bv.q[0] = *(const u32x4*)vp;
        bv.q[1] = *(const u32x4*)(vp + 16);
        O[dj] = wmma_bf16(ap.v, bv.v, O[dj]);
      }
    }
    __syncthreads();
  }

  // y[b, t, h*hd + d] = O / L   (bf16 for proj GEMM input)
#pragma unroll
  for (int dj = 0; dj < 4; ++dj)
#pragma unroll
    for (int r = 0; r < 8; ++r) {
      int row = q0 + wave * 16 + kh * 8 + r;
      float val = O[dj][r] / Lrow[r];
      y[((size_t)b * T + row) * C + h * hd + dj * 16 + lm] = f2bf(val);
    }
}

// ---------------------------------------------------------------------------
// u = bf16( silu(a) * b )   (in-place into a's buffer is fine)
// ---------------------------------------------------------------------------
__global__ void silu_mul_kernel(const unsigned short* __restrict__ a,
                                const unsigned short* __restrict__ b,
                                unsigned short* __restrict__ u, long long n) {
  long long i = (long long)blockIdx.x * blockDim.x + threadIdx.x;
  if (i < n) {
    float xv = bf2f(a[i]);
    float yv = bf2f(b[i]);
    float s = xv / (1.f + __expf(-xv));
    u[i] = f2bf(s * yv);
  }
}

// ---------------------------------------------------------------------------
// Host orchestration
// ---------------------------------------------------------------------------
extern "C" void kernel_launch(void* const* d_in, const int* in_sizes, int n_in,
                              void* d_out, int out_size, void* d_ws, size_t ws_size,
                              hipStream_t stream) {
  (void)in_sizes; (void)n_in; (void)out_size; (void)ws_size;
  const float* x      = (const float*)d_in[0];
  const float* w_attn = (const float*)d_in[1];
  const float* w_proj = (const float*)d_in[2];
  const float* w1     = (const float*)d_in[3];
  const float* w2     = (const float*)d_in[4];
  const float* w3     = (const float*)d_in[5];
  const float* g1     = (const float*)d_in[6];
  const float* g2     = (const float*)d_in[7];
  float* out = (float*)d_out;

  const int B = 4, T = 2048, C = 1024, H = 16, HID = 2730;
  const int BT = B * T;

  size_t off = 0;
  auto take = [&](size_t elems) -> unsigned short* {
    unsigned short* p = (unsigned short*)((char*)d_ws + off);
    off += ((elems * 2) + 255) & ~(size_t)255;
    return p;
  };
  unsigned short* h1b     = take((size_t)BT * C);
  unsigned short* h2b     = take((size_t)BT * C);
  unsigned short* wattn_t = take((size_t)3 * C * C);   // [3C, C]
  unsigned short* wproj_t = take((size_t)C * C);       // [C, C]
  unsigned short* w1t     = take((size_t)HID * C);     // [HID, C]
  unsigned short* w2t     = take((size_t)HID * C);     // [HID, C]
  unsigned short* w3t     = take((size_t)C * HID);     // [C, HID]
  unsigned short* qkvb    = take((size_t)BT * 3 * C);
  unsigned short* yb      = take((size_t)BT * C);
  unsigned short* a1b     = take((size_t)BT * HID);
  unsigned short* a2b     = take((size_t)BT * HID);

  const dim3 tcb(32, 8);
  // 1) weights -> bf16, transposed (so GEMM B staging is pure b128 copies)
  transpose_cast_kernel<<<dim3((3 * C + 31) / 32, (C + 31) / 32), tcb, 0, stream>>>(
      w_attn, wattn_t, C, 3 * C);
  transpose_cast_kernel<<<dim3((C + 31) / 32, (C + 31) / 32), tcb, 0, stream>>>(
      w_proj, wproj_t, C, C);
  transpose_cast_kernel<<<dim3((HID + 31) / 32, (C + 31) / 32), tcb, 0, stream>>>(
      w1, w1t, C, HID);
  transpose_cast_kernel<<<dim3((HID + 31) / 32, (C + 31) / 32), tcb, 0, stream>>>(
      w2, w2t, C, HID);
  transpose_cast_kernel<<<dim3((C + 31) / 32, (HID + 31) / 32), tcb, 0, stream>>>(
      w3, w3t, HID, C);

  // 2) rmsnorm (both gains) + seed out = x
  rmsnorm_dual_kernel<<<BT, 256, 0, stream>>>(x, g1, g2, h1b, h2b, out, C);

  // 3) qkv = ln1(x) @ w_attn    [8192 x 3072]
  gemm_bf16_kernel<0><<<dim3(3 * C / 128, BT / 128), 256, 0, stream>>>(
      h1b, wattn_t, qkvb, BT, 3 * C, C);

  // 4) flash attention -> y (bf16)
  attn_kernel<<<dim3(T / 64, B * H), 128, 0, stream>>>(qkvb, yb, T, C, H);

  // 5) out += y @ w_proj
  gemm_bf16_kernel<1><<<dim3(C / 128, BT / 128), 256, 0, stream>>>(
      yb, wproj_t, out, BT, C, C);

  // 6) a1 = ln2(x) @ w1 ; a2 = ln2(x) @ w2   [8192 x 2730]
  gemm_bf16_kernel<0><<<dim3((HID + 127) / 128, BT / 128), 256, 0, stream>>>(
      h2b, w1t, a1b, BT, HID, C);
  gemm_bf16_kernel<0><<<dim3((HID + 127) / 128, BT / 128), 256, 0, stream>>>(
      h2b, w2t, a2b, BT, HID, C);

  // 7) a1 = silu(a1) * a2 (in place)
  {
    long long n = (long long)BT * HID;
    silu_mul_kernel<<<(unsigned int)((n + 255) / 256), 256, 0, stream>>>(
        a1b, a2b, a1b, n);
  }

  // 8) out += a1 @ w3   (K = 2730, guarded 4B staging path)
  gemm_bf16_kernel<1><<<dim3(C / 128, BT / 128), 256, 0, stream>>>(
      a1b, w3t, out, BT, C, HID);
}
